// TemporalFusionTransformer_8426725834817
// MI455X (gfx1250) — matrix operand order repaired and checked by reading, not compiled, not run on hardware
//
#include <hip/hip_runtime.h>
#include <math.h>

#define BB 32
#define TT 256
#define DD 32
#define HH 128
#define NHEAD 4
#define HD (HH / NHEAD)      // 32
#define BT (BB * TT)         // 8192
#define BH (BB * NHEAD)      // 128

typedef __attribute__((ext_vector_type(16))) _Float16 v16h;
typedef __attribute__((ext_vector_type(8)))  float    v8f;

// ---------------------------------------------------------------------------
// WMMA fragment helpers (wave32, gfx1250 VGPR layouts per CDNA5 ISA 7.12.2)
// ---------------------------------------------------------------------------

static __device__ __forceinline__ v8f zero8() {
  v8f z;
#pragma unroll
  for (int i = 0; i < 8; ++i) z[i] = 0.f;
  return z;
}

// A: 16x32 f16 tile, LDS row-major [m][k] with stride lda.
// Lane L (L<16): row L, VGPR0..3 K=0..7, VGPR4..7 K=16..23.
// Lane L+16:     row L, VGPR0..3 K=8..15, VGPR4..7 K=24..31.
static __device__ __forceinline__ v16h frag_a(const _Float16* a, int lda) {
  const int lane = threadIdx.x & 31;
  const int m = lane & 15, hi = lane >> 4;
  v16h r;
#pragma unroll
  for (int e = 0; e < 16; ++e) {
    const int k = ((e >> 3) << 4) + (hi << 3) + (e & 7);
    r[e] = a[m * lda + k];
  }
  return r;
}

// B: 32x16 (KxN) f16 tile, staged in LDS as [n][k] with stride ldb.
// Lanes 0-15 hold K=0..7,16..23 of column n=lane; lanes 16-31 hold K=8..15,24..31.
static __device__ __forceinline__ v16h frag_b(const _Float16* b, int ldb) {
  const int lane = threadIdx.x & 31;
  const int n = lane & 15, hi = lane >> 4;
  v16h r;
#pragma unroll
  for (int e = 0; e < 16; ++e) {
    const int k = ((e >> 3) << 4) + (hi << 3) + (e & 7);
    r[e] = b[n * ldb + k];
  }
  return r;
}

static __device__ __forceinline__ v8f wmma32(v16h a, v16h b, v8f c) {
  // emits v_wmma_f32_16x16x32_f16
  return __builtin_amdgcn_wmma_f32_16x16x32_f16(false, a, false, b, (short)0, c,
                                                false, false);
}

static __device__ __forceinline__ float sigm(float x) { return 1.f / (1.f + __expf(-x)); }
static __device__ __forceinline__ float eluf(float x) { return x > 0.f ? x : (__expf(x) - 1.f); }
static __device__ __forceinline__ float act_apply(float v, int act) {
  if (act == 1) return eluf(v);
  if (act == 2) return sigm(v);
  return v;
}
static __device__ __forceinline__ float wave_sum(float v) {
#pragma unroll
  for (int o = 16; o > 0; o >>= 1) v += __shfl_xor(v, o);
  return v;
}
static __device__ __forceinline__ float wave_max(float v) {
#pragma unroll
  for (int o = 16; o > 0; o >>= 1) v = fmaxf(v, __shfl_xor(v, o));
  return v;
}

// ---------------------------------------------------------------------------
// Generic WMMA GEMM: Y[M,N] = act( X[M,K] @ W[N,K]^T + bias[N] )
// block=256 (8 waves), tile 64x128, K chunked by 128. fp32 io, f16 compute.
// ---------------------------------------------------------------------------
__global__ __launch_bounds__(256) void gemm_wmma_kernel(
    const float* __restrict__ X, const float* __restrict__ W,
    const float* __restrict__ bias, float* __restrict__ Y,
    int M, int N, int K, int act) {
  __shared__ _Float16 As[64 * 128];   // 16 KB
  __shared__ _Float16 Bs[128 * 128];  // 32 KB
  const int tid = threadIdx.x;
  const int m0 = blockIdx.x * 64;
  const int n0 = blockIdx.y * 128;
  const int wave = tid >> 5, lane = tid & 31;
  const int lncol = lane & 15, hi = lane >> 4;
  const int mtile = wave >> 1;
  const int ntb = (wave & 1) * 4;

  v8f acc[4];
#pragma unroll
  for (int j = 0; j < 4; ++j) acc[j] = zero8();

  for (int kc = 0; kc < K; kc += 128) {
    for (int i = tid; i < 64 * 128; i += 256) {
      int m = i >> 7, k = i & 127;
      int gr = m0 + m, gk = kc + k;
      As[i] = (_Float16)((gr < M && gk < K) ? X[(size_t)gr * K + gk] : 0.f);
    }
    for (int i = tid; i < 128 * 128; i += 256) {
      int n = i >> 7, k = i & 127;
      int gn = n0 + n, gk = kc + k;
      Bs[i] = (_Float16)((gn < N && gk < K) ? W[(size_t)gn * K + gk] : 0.f);
    }
    __syncthreads();
#pragma unroll
    for (int j = 0; j < 4; ++j) {
      const int ntile = ntb + j;
#pragma unroll
      for (int kk = 0; kk < 4; ++kk)
        acc[j] = wmma32(frag_a(As + mtile * 16 * 128 + kk * 32, 128),
                        frag_b(Bs + ntile * 16 * 128 + kk * 32, 128), acc[j]);
    }
    __syncthreads();
  }

#pragma unroll
  for (int j = 0; j < 4; ++j) {
    const int ntile = ntb + j;
#pragma unroll
    for (int v = 0; v < 8; ++v) {
      int gr = m0 + mtile * 16 + v + 8 * hi;
      int gc = n0 + ntile * 16 + lncol;
      if (gr < M && gc < N) {
        float val = acc[j][v] + (bias ? bias[gc] : 0.f);
        Y[(size_t)gr * N + gc] = act_apply(val, act);
      }
    }
  }
}

// ---------------------------------------------------------------------------
// Fused per-variable GRN + variable-selection.
// One block per 64 (b,t) rows; loops d=0..31, never materializes [B,T,D,H].
// Dynamic LDS: 128 KB.
// ---------------------------------------------------------------------------
__global__ __launch_bounds__(256) void var_grn_select_kernel(
    const float* __restrict__ x,
    const float* __restrict__ w1, const float* __restrict__ b1,
    const float* __restrict__ w2, const float* __restrict__ b2,
    const float* __restrict__ wg, const float* __restrict__ bg,
    const float* __restrict__ wsk, const float* __restrict__ bsk,
    const float* __restrict__ lnw, const float* __restrict__ lnb,
    const float* __restrict__ vsw, const float* __restrict__ vsb,
    float* __restrict__ sel) {
  extern __shared__ char smem[];
  _Float16* As = (_Float16*)smem;            // 64x128 f16
  _Float16* Bs = As + 64 * 128;              // 128x128 f16
  float* Ys = (float*)(Bs + 128 * 128);      // 64x128 f32
  float* sels = Ys + 64 * 128;               // 64x128 f32
  float* xts = sels + 64 * 128;              // 64x32 f32
  float* wsl = xts + 64 * 32;                // 64x32 f32

  const int tid = threadIdx.x;
  const int m0 = blockIdx.x * 64;
  const int wave = tid >> 5, lane = tid & 31;
  const int lncol = lane & 15, hi = lane >> 4;
  const int mtile = wave >> 1;
  const int ntb = (wave & 1) * 4;

  for (int i = tid; i < 64 * 32; i += 256) {
    int m = i >> 5, d = i & 31;
    xts[i] = x[(size_t)(m0 + m) * DD + d];
  }
  for (int i = tid; i < 64 * 128; i += 256) sels[i] = 0.f;
  __syncthreads();

  // variable-selection softmax weights (one row per thread)
  if (tid < 64) {
    float lg[DD];
    float mx = -1e30f;
    for (int dd = 0; dd < DD; ++dd) {
      float s = vsb[dd];
      for (int j = 0; j < DD; ++j) s += xts[tid * 32 + j] * vsw[dd * DD + j];
      lg[dd] = s;
      mx = fmaxf(mx, s);
    }
    float sum = 0.f;
    for (int dd = 0; dd < DD; ++dd) { lg[dd] = __expf(lg[dd] - mx); sum += lg[dd]; }
    float inv = 1.f / sum;
    for (int dd = 0; dd < DD; ++dd) wsl[tid * 32 + dd] = lg[dd] * inv;
  }
  __syncthreads();

  for (int d = 0; d < DD; ++d) {
    // h1 = elu(x*w1 + b1)  -> A tile (f16)
    for (int i = tid; i < 64 * 128; i += 256) {
      int m = i >> 7, h = i & 127;
      As[i] = (_Float16)eluf(xts[m * 32 + d] * w1[d * HH + h] + b1[d * HH + h]);
    }
    // B = w2[d] ([o][h] row-major == desired [n][k])
    for (int i = tid; i < 128 * 128; i += 256)
      Bs[i] = (_Float16)w2[(size_t)d * HH * HH + i];
    __syncthreads();

    v8f acc[4];
#pragma unroll
    for (int j = 0; j < 4; ++j) acc[j] = zero8();
#pragma unroll
    for (int j = 0; j < 4; ++j) {
      const int ntile = ntb + j;
#pragma unroll
      for (int kk = 0; kk < 4; ++kk)
        acc[j] = wmma32(frag_a(As + mtile * 16 * 128 + kk * 32, 128),
                        frag_b(Bs + ntile * 16 * 128 + kk * 32, 128), acc[j]);
    }
#pragma unroll
    for (int j = 0; j < 4; ++j) {
      const int ntile = ntb + j;
#pragma unroll
      for (int v = 0; v < 8; ++v) {
        int r = mtile * 16 + v + 8 * hi, c = ntile * 16 + lncol;
        Ys[r * 128 + c] = acc[j][v] + b2[d * HH + c];
      }
    }
    __syncthreads();

    // B = wg[d]
    for (int i = tid; i < 128 * 128; i += 256)
      Bs[i] = (_Float16)wg[(size_t)d * HH * HH + i];
    __syncthreads();

#pragma unroll
    for (int j = 0; j < 4; ++j) acc[j] = zero8();
#pragma unroll
    for (int j = 0; j < 4; ++j) {
      const int ntile = ntb + j;
#pragma unroll
      for (int kk = 0; kk < 4; ++kk)
        acc[j] = wmma32(frag_a(As + mtile * 16 * 128 + kk * 32, 128),
                        frag_b(Bs + ntile * 16 * 128 + kk * 32, 128), acc[j]);
    }
#pragma unroll
    for (int j = 0; j < 4; ++j) {
      const int ntile = ntb + j;
#pragma unroll
      for (int v = 0; v < 8; ++v) {
        int r = mtile * 16 + v + 8 * hi, c = ntile * 16 + lncol;
        float g = sigm(acc[j][v] + bg[d * HH + c]);
        float skip = xts[r * 32 + d] * wsk[d * HH + c] + bsk[d * HH + c];
        Ys[r * 128 + c] = g * Ys[r * 128 + c] + skip;
      }
    }
    __syncthreads();

    // per-row LayerNorm over H (64 rows, one per thread)
    if (tid < 64) {
      float mean = 0.f;
      for (int h = 0; h < HH; ++h) mean += Ys[tid * 128 + h];
      mean *= (1.f / 128.f);
      float var = 0.f;
      for (int h = 0; h < HH; ++h) { float dv = Ys[tid * 128 + h] - mean; var += dv * dv; }
      var *= (1.f / 128.f);
      float rstd = rsqrtf(var + 1e-5f);
      for (int h = 0; h < HH; ++h)
        Ys[tid * 128 + h] =
            (Ys[tid * 128 + h] - mean) * rstd * lnw[d * HH + h] + lnb[d * HH + h];
    }
    __syncthreads();

    // sel += w[b,t,d] * grn_out
    for (int i = tid; i < 64 * 128; i += 256) {
      int m = i >> 7;
      sels[i] += wsl[m * 32 + d] * Ys[i];
    }
    __syncthreads();
  }

  for (int i = tid; i < 64 * 128; i += 256) {
    int m = i >> 7, h = i & 127;
    sel[(size_t)(m0 + m) * HH + h] = sels[i];
  }
}

// ---------------------------------------------------------------------------
// LSTM recurrent scan: one persistent workgroup, whh pinned in LDS (f16).
// xg = x@wih^T + bih precomputed by gemm; bhh folded in here.
// Dynamic LDS: 216 KB (whh 128K + h 8K + c 16K + gates 64K), within 320 KB WGP.
// ---------------------------------------------------------------------------
__global__ __launch_bounds__(256) void lstm_scan_kernel(
    const float* __restrict__ xg, const float* __restrict__ whh,
    const float* __restrict__ bhh, float* __restrict__ out) {
  extern __shared__ char smem[];
  _Float16* Whs = (_Float16*)smem;        // [512][128] f16
  _Float16* Hs = Whs + 512 * 128;         // [32][128] f16
  float* Cs = (float*)(Hs + 32 * 128);    // [32][128] f32
  float* Gs = Cs + 32 * 128;              // [32][512] f32

  const int tid = threadIdx.x;
  const int wave = tid >> 5, lane = tid & 31;
  const int lncol = lane & 15, hi = lane >> 4;
  const int mtile = wave & 1;          // 2 M tiles (32 rows)
  const int nb = (wave >> 1) * 8;      // 8 N tiles per wave (32 total)

  for (int i = tid; i < 512 * 128; i += 256) Whs[i] = (_Float16)whh[i];
  for (int i = tid; i < 32 * 128; i += 256) { Hs[i] = (_Float16)0.f; Cs[i] = 0.f; }
  __syncthreads();

  for (int t = 0; t < TT; ++t) {
    // gates_rec = h @ whh^T  (M=32, N=512, K=128)
#pragma unroll
    for (int j = 0; j < 8; ++j) {
      const int ntile = nb + j;
      v8f acc = zero8();
#pragma unroll
      for (int kk = 0; kk < 4; ++kk)
        acc = wmma32(frag_a(Hs + mtile * 16 * 128 + kk * 32, 128),
                     frag_b(Whs + ntile * 16 * 128 + kk * 32, 128), acc);
#pragma unroll
      for (int v = 0; v < 8; ++v)
        Gs[(mtile * 16 + v + 8 * hi) * 512 + ntile * 16 + lncol] = acc[v];
    }
    __syncthreads();

    // cell update (PyTorch gate order i,f,g,o)
    for (int i = tid; i < 32 * 128; i += 256) {
      int m = i >> 7, h = i & 127;
      size_t ro = ((size_t)m * TT + t) * 512;
      float gi = xg[ro + h] + bhh[h] + Gs[m * 512 + h];
      float gf = xg[ro + 128 + h] + bhh[128 + h] + Gs[m * 512 + 128 + h];
      float gg = xg[ro + 256 + h] + bhh[256 + h] + Gs[m * 512 + 256 + h];
      float go = xg[ro + 384 + h] + bhh[384 + h] + Gs[m * 512 + 384 + h];
      float c = sigm(gf) * Cs[i] + sigm(gi) * tanhf(gg);
      Cs[i] = c;
      float hn = sigm(go) * tanhf(c);
      Hs[i] = (_Float16)hn;
      out[((size_t)m * TT + t) * HH + h] = hn;
    }
    __syncthreads();
  }
}

// ---------------------------------------------------------------------------
// Attention: scores = scale * Q K^T (K=32 -> one WMMA per 16x16 tile)
// ---------------------------------------------------------------------------
__global__ __launch_bounds__(256) void attn_scores_kernel(
    const float* __restrict__ q, const float* __restrict__ k, float* __restrict__ s) {
  __shared__ _Float16 Qs[64 * 32];
  __shared__ _Float16 Ks[128 * 32];
  const int tid = threadIdx.x;
  const int m0 = blockIdx.x * 64;
  const int n0 = blockIdx.y * 128;
  const int bh = blockIdx.z;
  const int b = bh >> 2, head = bh & 3;

  for (int i = tid; i < 64 * 32; i += 256) {
    int m = i >> 5, kk = i & 31;
    Qs[i] = (_Float16)q[((size_t)b * TT + m0 + m) * HH + head * HD + kk];
  }
  for (int i = tid; i < 128 * 32; i += 256) {
    int n = i >> 5, kk = i & 31;
    Ks[i] = (_Float16)k[((size_t)b * TT + n0 + n) * HH + head * HD + kk];
  }
  __syncthreads();

  const int wave = tid >> 5, lane = tid & 31;
  const int lncol = lane & 15, hi = lane >> 4;
  const int mtile = wave >> 1, ntb = (wave & 1) * 4;
  const float scale = 0.17677669529663687f;  // 1/sqrt(32)
#pragma unroll
  for (int j = 0; j < 4; ++j) {
    const int ntile = ntb + j;
    v8f acc = zero8();
    acc = wmma32(frag_a(Qs + mtile * 16 * 32, 32), frag_b(Ks + ntile * 16 * 32, 32), acc);
#pragma unroll
    for (int v = 0; v < 8; ++v) {
      int r = m0 + mtile * 16 + v + 8 * hi, c = n0 + ntile * 16 + lncol;
      s[((size_t)bh * TT + r) * TT + c] = acc[v] * scale;
    }
  }
}

__global__ __launch_bounds__(256) void softmax_kernel(float* __restrict__ s) {
  const int row = blockIdx.x * 8 + (threadIdx.x >> 5);
  const int lane = threadIdx.x & 31;
  float* r = s + (size_t)row * TT;
  float vals[8];
  float mx = -1e30f;
#pragma unroll
  for (int j = 0; j < 8; ++j) { vals[j] = r[lane + 32 * j]; mx = fmaxf(mx, vals[j]); }
  mx = wave_max(mx);
  float sum = 0.f;
#pragma unroll
  for (int j = 0; j < 8; ++j) { vals[j] = __expf(vals[j] - mx); sum += vals[j]; }
  sum = wave_sum(sum);
  float inv = 1.f / sum;
#pragma unroll
  for (int j = 0; j < 8; ++j) r[lane + 32 * j] = vals[j] * inv;
}

// ctx[b,t,head*32+n] = sum_s w[bh,t,s] * v[b,s,head*32+n]   (M=256,K=256,N=32)
__global__ __launch_bounds__(256) void attn_av_kernel(
    const float* __restrict__ s, const float* __restrict__ v, float* __restrict__ ctx) {
  __shared__ _Float16 As[64 * 128];
  __shared__ _Float16 Bs[32 * 128];
  const int tid = threadIdx.x;
  const int m0 = blockIdx.x * 64;
  const int bh = blockIdx.y;
  const int b = bh >> 2, head = bh & 3;
  const int wave = tid >> 5, lane = tid & 31;
  const int lncol = lane & 15, hi = lane >> 4;
  const int mtile = wave >> 1, ntile = wave & 1;

  v8f acc = zero8();
  for (int kc = 0; kc < TT; kc += 128) {
    for (int i = tid; i < 64 * 128; i += 256) {
      int m = i >> 7, kk = i & 127;
      As[i] = (_Float16)s[((size_t)bh * TT + m0 + m) * TT + kc + kk];
    }
    for (int i = tid; i < 32 * 128; i += 256) {
      int n = i >> 7, kk = i & 127;
      Bs[i] = (_Float16)v[((size_t)b * TT + kc + kk) * HH + head * HD + n];
    }
    __syncthreads();
#pragma unroll
    for (int kk = 0; kk < 4; ++kk)
      acc = wmma32(frag_a(As + mtile * 16 * 128 + kk * 32, 128),
                   frag_b(Bs + ntile * 16 * 128 + kk * 32, 128), acc);
    __syncthreads();
  }
#pragma unroll
  for (int vv = 0; vv < 8; ++vv) {
    int r = m0 + mtile * 16 + vv + 8 * hi, c = ntile * 16 + lncol;
    ctx[((size_t)b * TT + r) * HH + head * HD + c] = acc[vv];
  }
}

// ---------------------------------------------------------------------------
// Elementwise / normalization kernels
// ---------------------------------------------------------------------------
__global__ __launch_bounds__(256) void add_kernel(const float* __restrict__ a,
                                                  const float* __restrict__ b,
                                                  float* __restrict__ c, int n) {
  int i = blockIdx.x * 256 + threadIdx.x;
  if (i < n) c[i] = a[i] + b[i];
}

__global__ __launch_bounds__(256) void pe_add_kernel(float* __restrict__ h) {
  int i = blockIdx.x * 256 + threadIdx.x;
  if (i >= BT * HH) return;
  int hcol = i & 127;
  int t = (i >> 7) & (TT - 1);
  float div = __expf((float)((hcol >> 1) * 2) * (-9.210340371976184f / 128.f));
  float ang = (float)t * div;
  h[i] += (hcol & 1) ? __cosf(ang) : __sinf(ang);
}

// y = LN( (gate? gate*a : a) + skip ) * w + b, over H=128; wave per row
__global__ __launch_bounds__(256) void ln_gate_kernel(
    const float* __restrict__ a, const float* __restrict__ gate,
    const float* __restrict__ skip, const float* __restrict__ w,
    const float* __restrict__ b, float* __restrict__ y) {
  const int row = blockIdx.x * 8 + (threadIdx.x >> 5);
  const int lane = threadIdx.x & 31;
  const size_t base = (size_t)row * HH;
  float v[4];
  float sum = 0.f;
#pragma unroll
  for (int j = 0; j < 4; ++j) {
    int h = lane + 32 * j;
    float s = gate ? gate[base + h] * a[base + h] : a[base + h];
    s += skip[base + h];
    v[j] = s;
    sum += s;
  }
  float mean = wave_sum(sum) * (1.f / 128.f);
  float vs = 0.f;
#pragma unroll
  for (int j = 0; j < 4; ++j) { float dv = v[j] - mean; vs += dv * dv; }
  float var = wave_sum(vs) * (1.f / 128.f);
  float rstd = rsqrtf(var + 1e-5f);
#pragma unroll
  for (int j = 0; j < 4; ++j) {
    int h = lane + 32 * j;
    y[base + h] = (v[j] - mean) * rstd * w[h] + b[h];
  }
}

// out[b] = (dot(h[b, T-1, :], w) + bias) / temp
__global__ void final_out_kernel(const float* __restrict__ h, const float* __restrict__ w,
                                 const float* __restrict__ bo, const float* __restrict__ temp,
                                 float* __restrict__ out) {
  const int b = blockIdx.x;
  const int lane = threadIdx.x;
  const size_t base = ((size_t)b * TT + (TT - 1)) * HH;
  float s = 0.f;
#pragma unroll
  for (int j = 0; j < 4; ++j) { int hh = lane + 32 * j; s += h[base + hh] * w[hh]; }
  s = wave_sum(s);
  if (lane == 0) out[b] = (s + bo[0]) / temp[0];
}

// ---------------------------------------------------------------------------
// Host orchestration
// ---------------------------------------------------------------------------
extern "C" void kernel_launch(void* const* d_in, const int* in_sizes, int n_in,
                              void* d_out, int out_size, void* d_ws, size_t ws_size,
                              hipStream_t stream) {
  (void)in_sizes; (void)n_in; (void)out_size; (void)ws_size;
  const float** in = (const float**)d_in;
  const float* x = in[0];
  const float *vg_w1 = in[1], *vg_b1 = in[2], *vg_w2 = in[3], *vg_b2 = in[4],
              *vg_wg = in[5], *vg_bg = in[6], *vg_ws = in[7], *vg_bs = in[8],
              *vg_lnw = in[9], *vg_lnb = in[10];
  const float *vs_w = in[11], *vs_b = in[12];
  const float *l0_wih = in[13], *l0_whh = in[14], *l0_bih = in[15], *l0_bhh = in[16];
  const float *l1_wih = in[17], *l1_whh = in[18], *l1_bih = in[19], *l1_bhh = in[20];
  const float *ag_w1 = in[21], *ag_b1 = in[22], *ag_w2 = in[23], *ag_b2 = in[24],
              *ag_wg = in[25], *ag_bg = in[26], *ag_lnw = in[27], *ag_lnb = in[28];
  const float *wq = in[29], *bq = in[30], *wk = in[31], *bk = in[32],
              *wv = in[33], *bv = in[34], *wo = in[35], *bo = in[36];
  const float *pg_w1 = in[37], *pg_b1 = in[38], *pg_w2 = in[39], *pg_b2 = in[40],
              *pg_wg = in[41], *pg_bg = in[42], *pg_lnw = in[43], *pg_lnb = in[44];
  const float *ff_w1 = in[45], *ff_b1 = in[46], *ff_w2 = in[47], *ff_b2 = in[48],
              *ff_wg = in[49], *ff_bg = in[50], *ff_lnw = in[51], *ff_lnb = in[52];
  const float *nw = in[53], *nb_ = in[54];
  const float *ow = in[55], *ob = in[56], *temp = in[57];

  float* ws = (float*)d_ws;
  const size_t F = (size_t)BT * HH;  // 1,048,576 floats
  float* sel = ws + 0 * F;
  float* t0 = ws + 1 * F;
  float* t1 = ws + 2 * F;
  float* xg = ws + 3 * F;        // 4F ([BT,512])
  float* attn_in = ws + 7 * F;
  float* h1b = ws + 8 * F;       // 2F (ff_grn: [BT,256])
  float* h2b = ws + 10 * F;
  float* gb = ws + 11 * F;
  float* qb = ws + 12 * F;
  float* kb = ws + 13 * F;
  float* vb = ws + 14 * F;
  float* sc = ws + 15 * F;       // 8F ([BH,256,256])
  float* ctx = ws + 23 * F;
  float* aout = ws + 24 * F;
  float* pin = ws + 25 * F;
  float* post = ws + 26 * F;
  float* ffb = ws + 27 * F;
  float* lnf = ws + 28 * F;

  auto gemm = [&](const float* X, const float* W, const float* bias, float* Y,
                  int M, int N, int K, int act) {
    dim3 g((M + 63) / 64, (N + 127) / 128);
    gemm_wmma_kernel<<<g, 256, 0, stream>>>(X, W, bias, Y, M, N, K, act);
  };

  const size_t VAR_SMEM =
      (64 * 128 + 128 * 128) * sizeof(_Float16) +
      (64 * 128 * 2 + 64 * 32 * 2) * sizeof(float);               // 128 KB
  const size_t SCAN_SMEM =
      (512 * 128 + 32 * 128) * sizeof(_Float16) +
      (32 * 128 + 32 * 512) * sizeof(float);                      // 216 KB

  // 1) fused per-variable GRNs + variable selection -> sel[B,T,H]
  var_grn_select_kernel<<<BT / 64, 256, VAR_SMEM, stream>>>(
      x, vg_w1, vg_b1, vg_w2, vg_b2, vg_wg, vg_bg, vg_ws, vg_bs, vg_lnw, vg_lnb,
      vs_w, vs_b, sel);

  // 2) LSTM layer 0
  gemm(sel, l0_wih, l0_bih, xg, BT, 4 * HH, HH, 0);
  lstm_scan_kernel<<<1, 256, SCAN_SMEM, stream>>>(xg, l0_whh, l0_bhh, t0);
  // 3) LSTM layer 1
  gemm(t0, l1_wih, l1_bih, xg, BT, 4 * HH, HH, 0);
  lstm_scan_kernel<<<1, 256, SCAN_SMEM, stream>>>(xg, l1_whh, l1_bhh, t1);

  // 4) sinusoidal positional encoding
  pe_add_kernel<<<(BT * HH) / 256, 256, 0, stream>>>(t1);

  // 5) attn_grn
  gemm(t1, ag_w1, ag_b1, h1b, BT, HH, HH, 1);
  gemm(h1b, ag_w2, ag_b2, h2b, BT, HH, HH, 0);
  gemm(h1b, ag_wg, ag_bg, gb, BT, HH, HH, 2);
  ln_gate_kernel<<<BT / 8, 256, 0, stream>>>(h2b, gb, t1, ag_lnw, ag_lnb, attn_in);

  // 6) attention
  gemm(attn_in, wq, bq, qb, BT, HH, HH, 0);
  gemm(attn_in, wk, bk, kb, BT, HH, HH, 0);
  gemm(attn_in, wv, bv, vb, BT, HH, HH, 0);
  attn_scores_kernel<<<dim3(TT / 64, TT / 128, BH), 256, 0, stream>>>(qb, kb, sc);
  softmax_kernel<<<(BH * TT) / 8, 256, 0, stream>>>(sc);
  attn_av_kernel<<<dim3(TT / 64, BH), 256, 0, stream>>>(sc, vb, ctx);
  gemm(ctx, wo, bo, aout, BT, HH, HH, 0);

  // 7) post_grn on (attn_out + attn_in)
  add_kernel<<<(BT * HH) / 256, 256, 0, stream>>>(aout, attn_in, pin, BT * HH);
  gemm(pin, pg_w1, pg_b1, h1b, BT, HH, HH, 1);
  gemm(h1b, pg_w2, pg_b2, h2b, BT, HH, HH, 0);
  gemm(h1b, pg_wg, pg_bg, gb, BT, HH, HH, 2);
  ln_gate_kernel<<<BT / 8, 256, 0, stream>>>(h2b, gb, pin, pg_lnw, pg_lnb, post);

  // 8) ff_grn (hidden 2H)
  gemm(post, ff_w1, ff_b1, h1b, BT, 2 * HH, HH, 1);
  gemm(h1b, ff_w2, ff_b2, h2b, BT, HH, 2 * HH, 0);
  gemm(h1b, ff_wg, ff_bg, gb, BT, HH, 2 * HH, 2);
  ln_gate_kernel<<<BT / 8, 256, 0, stream>>>(h2b, gb, post, ff_lnw, ff_lnb, ffb);

  // 9) final LN(ff + post), last timestep projection, temperature
  ln_gate_kernel<<<BT / 8, 256, 0, stream>>>(ffb, nullptr, post, nw, nb_, lnf);
  final_out_kernel<<<BB, 32, 0, stream>>>(lnf, ow, ob, temp, (float*)d_out);
}
